// mLSTMCell_42949673565
// MI455X (gfx1250) — compile-verified, matrix operand order checked by reading, not run on hardware
//
#include <hip/hip_runtime.h>
#include <hip/hip_bf16.h>

// mLSTM cell forward for MI455X (gfx1250), wave32 + WMMA bf16.
// B=2, S=2048, E=1024, NH=4, HD=256.

constexpr int B  = 2;
constexpr int S  = 2048;
constexpr int E  = 1024;
constexpr int NH = 4;
constexpr int HD = 256;         // E / NH
constexpr float SCALE = 0.0625f; // 1/sqrt(256)

typedef __attribute__((ext_vector_type(16))) __bf16 v16bf;
typedef __attribute__((ext_vector_type(8)))  float  v8f;

__device__ __forceinline__ __bf16 f2bf(float f) {
  unsigned u = __builtin_bit_cast(unsigned, f);
  unsigned r = (u + 0x7FFFu + ((u >> 16) & 1u)) >> 16;   // round-to-nearest-even
  return __builtin_bit_cast(__bf16, (unsigned short)r);
}

__device__ __forceinline__ float logsig(float x) {
  // log(sigmoid(x)) = -softplus(-x), numerically stable
  float ax = fabsf(x);
  float t  = __logf(1.0f + __expf(-ax));
  return (x >= 0.0f) ? -t : (x - t);
}

// ---------------------------------------------------------------------------
// Kernel 1: gate projections.  ig/fg[b,h,s] = concat(q,k,v)[b,s,:] @ W[:,h] + bias
// grid = B*S blocks, 256 threads
// ---------------------------------------------------------------------------
__global__ void gates_kernel(const float* __restrict__ q,
                             const float* __restrict__ k,
                             const float* __restrict__ v,
                             const float* __restrict__ igk, const float* __restrict__ igb,
                             const float* __restrict__ fgk, const float* __restrict__ fgb,
                             float* __restrict__ ig_ws, float* __restrict__ fg_ws) {
  int bs = blockIdx.x;            // b*S + s
  int b  = bs / S, s = bs % S;
  const float* qr = q + (size_t)bs * E;
  const float* kr = k + (size_t)bs * E;
  const float* vr = v + (size_t)bs * E;

  float accI[NH] = {0.f, 0.f, 0.f, 0.f};
  float accF[NH] = {0.f, 0.f, 0.f, 0.f};
  for (int e = threadIdx.x; e < 3 * E; e += blockDim.x) {
    float x = (e < E) ? qr[e] : (e < 2 * E) ? kr[e - E] : vr[e - 2 * E];
#pragma unroll
    for (int h = 0; h < NH; ++h) {
      accI[h] += x * igk[(size_t)e * NH + h];
      accF[h] += x * fgk[(size_t)e * NH + h];
    }
  }

  __shared__ float redI[256 * NH];
  __shared__ float redF[256 * NH];
  int tid = threadIdx.x;
#pragma unroll
  for (int h = 0; h < NH; ++h) { redI[tid * NH + h] = accI[h]; redF[tid * NH + h] = accF[h]; }
  __syncthreads();
  for (int off = 128; off > 0; off >>= 1) {
    if (tid < off) {
#pragma unroll
      for (int h = 0; h < NH; ++h) {
        redI[tid * NH + h] += redI[(tid + off) * NH + h];
        redF[tid * NH + h] += redF[(tid + off) * NH + h];
      }
    }
    __syncthreads();
  }
  if (tid < NH) {
    ig_ws[((size_t)b * NH + tid) * S + s] = redI[tid] + igb[tid];
    fg_ws[((size_t)b * NH + tid) * S + s] = redF[tid] + fgb[tid];
  }
}

// ---------------------------------------------------------------------------
// Kernel 2: per-(b,h) scan.  cs_t = cumsum logsigmoid(fg);  b_t = ig_t - cs_t;
// maxb_t = prefix-max of b.   grid = B*NH blocks of 32 (lane 0 works).
// ---------------------------------------------------------------------------
__global__ void scan_kernel(const float* __restrict__ ig_ws,
                            const float* __restrict__ fg_ws,
                            float* __restrict__ bv_ws,
                            float* __restrict__ maxb_ws,
                            float* __restrict__ cs_ws) {
  int bh = blockIdx.x;
  if (threadIdx.x != 0) return;
  const float* ig = ig_ws + (size_t)bh * S;
  const float* fg = fg_ws + (size_t)bh * S;
  float cs = 0.0f, mb = -3.0e38f;
  for (int t = 0; t < S; ++t) {
    cs += logsig(fg[t]);
    float bt = ig[t] - cs;
    mb = fmaxf(mb, bt);
    cs_ws  [(size_t)bh * S + t] = cs;
    bv_ws  [(size_t)bh * S + t] = bt;
    maxb_ws[(size_t)bh * S + t] = mb;
  }
}

// ---------------------------------------------------------------------------
// Kernel 3: convert K -> bf16 (same [b,s,e] layout) and V -> bf16 transposed
// to [b,h,d,s] so P@V B-fragments load contiguously along s.
// ---------------------------------------------------------------------------
__global__ void cvt_kernel(const float* __restrict__ k,
                           const float* __restrict__ v,
                           __bf16* __restrict__ kbf,
                           __bf16* __restrict__ vt) {
  size_t idx = (size_t)blockIdx.x * 256 + threadIdx.x;  // over B*S*E
  if (idx >= (size_t)B * S * E) return;
  kbf[idx] = f2bf(k[idx]);
  size_t e  = idx % E;
  size_t bs = idx / E;
  size_t s  = bs % S;
  size_t b  = bs / S;
  size_t h  = e / HD, d = e % HD;
  vt[(((b * NH + h) * HD) + d) * S + s] = f2bf(v[idx]);
}

// ---------------------------------------------------------------------------
// Kernel 4: causal decay-attention + normalizer + fused RMSNorm.
// Block = 128 threads (4 waves).  Wave w owns 16 rows; block owns 64 rows.
// grid = (S/64, B*NH).
//
// Fragment layouts per CDNA5 ISA 7.12.2 (wave32):
//   A (16x32 bf16): lane L -> row M = L&15; element e -> K = ((e>>3)<<4) + ((L>>4)<<3) + (e&7)
//   B (32x16 bf16): lane L -> col N = L&15; same K mapping as A (mirrored)
//   C/D (16x16 f32): lane L -> col N = L&15; VGPR r -> row M = r + 8*(L>>4)
// ---------------------------------------------------------------------------
__launch_bounds__(128)
__global__ void mlstm_attn(const float*  __restrict__ q,
                           const __bf16* __restrict__ kbf,
                           const __bf16* __restrict__ vt,
                           const float*  __restrict__ bv_ws,
                           const float*  __restrict__ maxb_ws,
                           const float*  __restrict__ cs_ws,
                           const float*  __restrict__ rms_scale,
                           float* __restrict__ out) {
  const int tile = blockIdx.x;          // row tile (64 rows)
  const int bh   = blockIdx.y;          // b*NH + h
  const int b    = bh / NH, h = bh % NH;
  const int wave = threadIdx.x >> 5;
  const int lane = threadIdx.x & 31;
  const int n    = lane & 15;           // N / M-within-fragment index
  const int hi   = lane >> 4;           // lane half
  const int r0   = tile * 64 + wave * 16;  // this wave's first row

  __shared__ __align__(16) unsigned short plds[4][16 * 32];  // per-wave P repack

  const v8f vzero = {0.f, 0.f, 0.f, 0.f, 0.f, 0.f, 0.f, 0.f};

  // ---- load Q fragments (A layout), converting fp32 -> bf16 on the fly ----
  const float* qrow = q + ((size_t)b * S + (r0 + n)) * E + (size_t)h * HD;
  v16bf qf[8];
#pragma unroll
  for (int kk = 0; kk < 8; ++kk) {
    const int kb = kk * 32 + hi * 8;
#pragma unroll
    for (int e = 0; e < 8; ++e) {
      qf[kk][e]     = f2bf(qrow[kb + e]);
      qf[kk][e + 8] = f2bf(qrow[kb + 16 + e]);
    }
  }

  // per-row constants for this lane's 8 rows (M = r + 8*hi)
  const float* maxbp = maxb_ws + (size_t)bh * S;
  float mbl[8];
#pragma unroll
  for (int r = 0; r < 8; ++r) mbl[r] = maxbp[r0 + r + 8 * hi];

  v8f acc[16];
#pragma unroll
  for (int t = 0; t < 16; ++t) acc[t] = vzero;
  float rs[8] = {0.f, 0.f, 0.f, 0.f, 0.f, 0.f, 0.f, 0.f};

  const __bf16* kbase = kbf + ((size_t)b * S) * E + (size_t)h * HD;
  const __bf16* vtb   = vt + (size_t)bh * HD * S;
  const float*  bvp   = bv_ws + (size_t)bh * S;

  for (int jb = 0; jb < r0 + 16; jb += 32) {
    // ---------------- scores: Q (16xHD) @ K^T -> 16x32 ----------------
    v8f s0 = vzero, s1 = vzero;
    {
      const __bf16* krow0 = kbase + (size_t)(jb + n) * E;
      const __bf16* krow1 = kbase + (size_t)(jb + 16 + n) * E;
#pragma unroll
      for (int kk = 0; kk < 8; ++kk) {
        const int kb = kk * 32 + hi * 8;
        v16bf kf0, kf1;
#pragma unroll
        for (int e = 0; e < 8; ++e) {
          kf0[e]     = krow0[kb + e];
          kf0[e + 8] = krow0[kb + 16 + e];
          kf1[e]     = krow1[kb + e];
          kf1[e + 8] = krow1[kb + 16 + e];
        }
        s0 = __builtin_amdgcn_wmma_f32_16x16x32_bf16(false, qf[kk], false, kf0,
                                                     (short)0, s0, false, false);
        s1 = __builtin_amdgcn_wmma_f32_16x16x32_bf16(false, qf[kk], false, kf1,
                                                     (short)0, s1, false, false);
      }
    }

    // ------------- decay / mask / row-sum, repack P to LDS -------------
    const float bj0 = bvp[jb + n];
    const float bj1 = bvp[jb + 16 + n];
#pragma unroll
    for (int r = 0; r < 8; ++r) {
      const int i = r0 + r + 8 * hi;
      const float d0 = (jb + n      <= i) ? __expf(bj0 - mbl[r]) : 0.0f;
      const float d1 = (jb + 16 + n <= i) ? __expf(bj1 - mbl[r]) : 0.0f;
      const float c0 = s0[r] * SCALE * d0;
      const float c1 = s1[r] * SCALE * d1;
      rs[r] += c0 + c1;
      const int m = r + 8 * hi;
      plds[wave][m * 32 + n]      = __builtin_bit_cast(unsigned short, f2bf(c0));
      plds[wave][m * 32 + 16 + n] = __builtin_bit_cast(unsigned short, f2bf(c1));
    }
    // same-wave LDS RAW: DS ops are in-order per wave; fence the compiler + DScnt
    asm volatile("s_wait_dscnt 0" ::: "memory");

    // read P back in A layout (16x32 bf16)
    v16bf pf;
#pragma unroll
    for (int e = 0; e < 8; ++e) {
      pf[e]     = __builtin_bit_cast(__bf16, plds[wave][n * 32 + hi * 8 + e]);
      pf[e + 8] = __builtin_bit_cast(__bf16, plds[wave][n * 32 + 16 + hi * 8 + e]);
    }
    asm volatile("s_wait_dscnt 0" ::: "memory");

    // ---------------- P (16x32) @ V (32xHD) accumulate ----------------
#pragma unroll
    for (int t = 0; t < 16; ++t) {
      const __bf16* vrow = vtb + (size_t)(t * 16 + n) * S + jb;
      v16bf vf;
#pragma unroll
      for (int e = 0; e < 8; ++e) {
        vf[e]     = vrow[hi * 8 + e];
        vf[e + 8] = vrow[16 + hi * 8 + e];
      }
      acc[t] = __builtin_amdgcn_wmma_f32_16x16x32_bf16(false, pf, false, vf,
                                                       (short)0, acc[t], false, false);
    }
  }

  // ---------------- row-sum reduction across the 16 lanes of each half ----
#pragma unroll
  for (int off = 1; off < 16; off <<= 1) {
#pragma unroll
    for (int r = 0; r < 8; ++r) rs[r] += __shfl_xor(rs[r], off, 32);
  }

  // denominator n_i = max(rowsum, exp(-cs_i - maxb_i)); divide accumulators
  const float* csp = cs_ws + (size_t)bh * S;
  float inv[8];
#pragma unroll
  for (int r = 0; r < 8; ++r) {
    const int i = r0 + r + 8 * hi;
    const float nden = fmaxf(rs[r], __expf(-csp[i] - mbl[r]));
    inv[r] = 1.0f / (nden + 1e-6f);
  }

  // fused RMSNorm over head dim (256) with (1 + rms_scale)
  float ssq[8] = {0.f, 0.f, 0.f, 0.f, 0.f, 0.f, 0.f, 0.f};
#pragma unroll
  for (int t = 0; t < 16; ++t) {
#pragma unroll
    for (int r = 0; r < 8; ++r) {
      const float hv = acc[t][r] * inv[r];
      acc[t][r] = hv;
      ssq[r] += hv * hv;
    }
  }
#pragma unroll
  for (int off = 1; off < 16; off <<= 1) {
#pragma unroll
    for (int r = 0; r < 8; ++r) ssq[r] += __shfl_xor(ssq[r], off, 32);
  }
  float rnorm[8];
#pragma unroll
  for (int r = 0; r < 8; ++r) rnorm[r] = rsqrtf(ssq[r] * (1.0f / HD) + 1e-6f);

  // write out[b, i, h*HD + col]
  float* obase = out + ((size_t)b * S) * E + (size_t)h * HD;
#pragma unroll
  for (int t = 0; t < 16; ++t) {
    const int col = t * 16 + n;
    const float g = 1.0f + rms_scale[col];
#pragma unroll
    for (int r = 0; r < 8; ++r) {
      const int i = r0 + r + 8 * hi;
      obase[(size_t)i * E + col] = acc[t][r] * rnorm[r] * g;
    }
  }
}

// ---------------------------------------------------------------------------
extern "C" void kernel_launch(void* const* d_in, const int* in_sizes, int n_in,
                              void* d_out, int out_size, void* d_ws, size_t ws_size,
                              hipStream_t stream) {
  const float* q   = (const float*)d_in[0];
  const float* k   = (const float*)d_in[1];
  const float* v   = (const float*)d_in[2];
  const float* igk = (const float*)d_in[3];
  const float* igb = (const float*)d_in[4];
  const float* fgk = (const float*)d_in[5];
  const float* fgb = (const float*)d_in[6];
  const float* rms = (const float*)d_in[7];
  float* out = (float*)d_out;

  char* ws = (char*)d_ws;
  const size_t BNS = (size_t)B * NH * S;
  float* ig_ws   = (float*)ws; ws += BNS * sizeof(float);
  float* fg_ws   = (float*)ws; ws += BNS * sizeof(float);
  float* cs_ws   = (float*)ws; ws += BNS * sizeof(float);
  float* bv_ws   = (float*)ws; ws += BNS * sizeof(float);
  float* maxb_ws = (float*)ws; ws += BNS * sizeof(float);
  __bf16* kbf = (__bf16*)ws; ws += (size_t)B * S * E * sizeof(__bf16);
  __bf16* vt  = (__bf16*)ws; ws += (size_t)B * S * E * sizeof(__bf16);

  gates_kernel<<<B * S, 256, 0, stream>>>(q, k, v, igk, igb, fgk, fgb, ig_ws, fg_ws);
  scan_kernel<<<B * NH, 32, 0, stream>>>(ig_ws, fg_ws, bv_ws, maxb_ws, cs_ws);
  cvt_kernel<<<(B * S * E) / 256, 256, 0, stream>>>(k, v, kbf, vt);
  dim3 grid(S / 64, B * NH);
  mlstm_attn<<<grid, 128, 0, stream>>>(q, kbf, vt, bv_ws, maxb_ws, cs_ws, rms, out);
}
